// BasicBlock_HMU_49520972923445
// MI455X (gfx1250) — compile-verified, matrix operand order checked by reading, not run on hardware
//
#include <hip/hip_runtime.h>
#include <hip/hip_bf16.h>

// ---------------------------------------------------------------------------
// CDNA5 (gfx1250) fused HMU-block. bf16 WMMA (v_wmma_f32_16x16x32_bf16),
// wave32. Each wave computes a 32(batch) x 16(n) output tile: two M-tiles
// share the 5 B fragments (mu-dot + 4 v-dots; v pre-transposed to (k,n,d) so
// all C-fragments share the lane->n mapping and the quadratic form folds
// elementwise in registers). 14 b128 loads per 10 WMMAs per 32-K step.
// ---------------------------------------------------------------------------

typedef __attribute__((ext_vector_type(16))) __bf16 v16bf;
typedef __attribute__((ext_vector_type(8)))  float  v8f;

#define BN_EPS 1e-5f

// --- 16-bit A/B fragment load for 16x32 tile -------------------------------
// Layout (CDNA5 ISA, 16-bit A 16x32): lane L -> row/col = L%16,
// VGPR0..3 hold K = (L/16)*8 .. +7, VGPR4..7 hold K = 16 + (L/16)*8 .. +7.
// => two contiguous 16-byte loads per lane at k-offsets (L>>4)*8 and +16.
__device__ __forceinline__ v16bf load_frag(const __bf16* __restrict__ base,
                                           int ld, int k0, int lane) {
  const __bf16* p = base + (size_t)(lane & 15) * ld + k0 + ((lane >> 4) << 3);
  union { v16bf v; uint4 q[2]; } u;
  u.q[0] = *(const uint4*)(p);
  u.q[1] = *(const uint4*)(p + 16);
  return u.v;
}

// --- fused HMU GEMM: h_pre[b,n] = exp(-quad/d) -----------------------------
// grid = (N/64, B/32), block = 128 (4 waves); wave w handles n-tile
// n0 = blockIdx.x*64 + w*16, batch tiles b0 = blockIdx.y*32 (+0 and +16).
__global__ __launch_bounds__(128) void hmu_wmma(
    const __bf16* __restrict__ X,    // B x D (bf16, row-major)
    const __bf16* __restrict__ MU,   // N x D
    const __bf16* __restrict__ VT,   // K x N x D  (k-major transpose of v)
    const float*  __restrict__ lam,  // N
    const float*  __restrict__ mumu, // N  : ||mu_n||^2
    const float*  __restrict__ vmu,  // K x N : v_{n,k} . mu_n
    const float*  __restrict__ xx,   // B  : ||x_b||^2
    float* __restrict__ hpre,        // B x N
    int D, int N)
{
  const int lane = threadIdx.x & 31;
  const int wave = threadIdx.x >> 5;
  const int n0 = blockIdx.x * 64 + wave * 16;
  const int b0 = blockIdx.y * 32;

  const __bf16* Xb0 = X  + (size_t)b0 * D;
  const __bf16* Xb1 = X  + (size_t)(b0 + 16) * D;
  const __bf16* Mb  = MU + (size_t)n0 * D;
  const __bf16* V0  = VT + ((size_t)0 * N + n0) * D;
  const __bf16* V1  = VT + ((size_t)1 * N + n0) * D;
  const __bf16* V2  = VT + ((size_t)2 * N + n0) * D;
  const __bf16* V3  = VT + ((size_t)3 * N + n0) * D;

  v8f acc_mu0 = {}, acc_v00 = {}, acc_v10 = {}, acc_v20 = {}, acc_v30 = {};
  v8f acc_mu1 = {}, acc_v01 = {}, acc_v11 = {}, acc_v21 = {}, acc_v31 = {};

  for (int k0 = 0; k0 < D; k0 += 32) {
    v16bf a0  = load_frag(Xb0, D, k0, lane);
    v16bf a1  = load_frag(Xb1, D, k0, lane);
    v16bf fm  = load_frag(Mb,  D, k0, lane);
    v16bf f0  = load_frag(V0,  D, k0, lane);
    v16bf f1  = load_frag(V1,  D, k0, lane);
    v16bf f2  = load_frag(V2,  D, k0, lane);
    v16bf f3  = load_frag(V3,  D, k0, lane);
    acc_mu0 = __builtin_amdgcn_wmma_f32_16x16x32_bf16(false, a0, false, fm,
                                                      (short)0, acc_mu0, false, false);
    acc_mu1 = __builtin_amdgcn_wmma_f32_16x16x32_bf16(false, a1, false, fm,
                                                      (short)0, acc_mu1, false, false);
    acc_v00 = __builtin_amdgcn_wmma_f32_16x16x32_bf16(false, a0, false, f0,
                                                      (short)0, acc_v00, false, false);
    acc_v01 = __builtin_amdgcn_wmma_f32_16x16x32_bf16(false, a1, false, f0,
                                                      (short)0, acc_v01, false, false);
    acc_v10 = __builtin_amdgcn_wmma_f32_16x16x32_bf16(false, a0, false, f1,
                                                      (short)0, acc_v10, false, false);
    acc_v11 = __builtin_amdgcn_wmma_f32_16x16x32_bf16(false, a1, false, f1,
                                                      (short)0, acc_v11, false, false);
    acc_v20 = __builtin_amdgcn_wmma_f32_16x16x32_bf16(false, a0, false, f2,
                                                      (short)0, acc_v20, false, false);
    acc_v21 = __builtin_amdgcn_wmma_f32_16x16x32_bf16(false, a1, false, f2,
                                                      (short)0, acc_v21, false, false);
    acc_v30 = __builtin_amdgcn_wmma_f32_16x16x32_bf16(false, a0, false, f3,
                                                      (short)0, acc_v30, false, false);
    acc_v31 = __builtin_amdgcn_wmma_f32_16x16x32_bf16(false, a1, false, f3,
                                                      (short)0, acc_v31, false, false);
  }

  // C layout: VGPR r, lanes 0-15 -> M=r, lanes 16-31 -> M=r+8; N = lane%16.
  const int   col  = n0 + (lane & 15);
  const float lamn = lam[col];
  const float mm   = mumu[col];
  const float vm0  = vmu[0 * N + col];
  const float vm1  = vmu[1 * N + col];
  const float vm2  = vmu[2 * N + col];
  const float vm3  = vmu[3 * N + col];
  const float invd = 1.0f / (float)D;
  const int rb0 = b0 + ((lane >> 4) << 3);
#pragma unroll
  for (int r = 0; r < 8; ++r) {
    const int row = rb0 + r;
    const float sq = xx[row] - 2.0f * acc_mu0[r] + mm;
    const float p0 = acc_v00[r] - vm0;
    const float p1 = acc_v10[r] - vm1;
    const float p2 = acc_v20[r] - vm2;
    const float p3 = acc_v30[r] - vm3;
    const float quad = lamn * sq + p0 * p0 + p1 * p1 + p2 * p2 + p3 * p3;
    hpre[(size_t)row * N + col] = __expf(-quad * invd);
  }
#pragma unroll
  for (int r = 0; r < 8; ++r) {
    const int row = rb0 + 16 + r;
    const float sq = xx[row] - 2.0f * acc_mu1[r] + mm;
    const float p0 = acc_v01[r] - vm0;
    const float p1 = acc_v11[r] - vm1;
    const float p2 = acc_v21[r] - vm2;
    const float p3 = acc_v31[r] - vm3;
    const float quad = lamn * sq + p0 * p0 + p1 * p1 + p2 * p2 + p3 * p3;
    hpre[(size_t)row * N + col] = __expf(-quad * invd);
  }
}

// --- zero scratch (BN stats must be re-zeroed every call) ------------------
__global__ void zero_f32(float* __restrict__ p, int count) {
  int i = blockIdx.x * blockDim.x + threadIdx.x;
  if (i < count) p[i] = 0.0f;
}

// --- x -> bf16 + per-row ||x||^2 ------------------------------------------
__global__ __launch_bounds__(256) void prep_x(const float* __restrict__ x,
                                              __bf16* __restrict__ xbf,
                                              float* __restrict__ xx, int D) {
  const int row = blockIdx.x;
  __shared__ float red[256];
  float acc = 0.0f;
  for (int c = threadIdx.x; c < D; c += 256) {
    const float v = x[(size_t)row * D + c];
    xbf[(size_t)row * D + c] = (__bf16)v;
    acc += v * v;
  }
  red[threadIdx.x] = acc; __syncthreads();
  for (int s = 128; s > 0; s >>= 1) {
    if (threadIdx.x < s) red[threadIdx.x] += red[threadIdx.x + s];
    __syncthreads();
  }
  if (threadIdx.x == 0) xx[row] = red[0];
}

// --- per-n parameter prep: mu->bf16, ||mu||^2, v->(k,n,d) bf16, v.mu -------
__global__ __launch_bounds__(256) void prep_params(
    const float* __restrict__ mu, const float* __restrict__ v,
    __bf16* __restrict__ mubf, __bf16* __restrict__ vT,
    float* __restrict__ mumu, float* __restrict__ vmu, int N, int D) {
  const int nn = blockIdx.x;
  __shared__ float red[5][256];
  float mm = 0.0f, vm0 = 0.0f, vm1 = 0.0f, vm2 = 0.0f, vm3 = 0.0f;
  for (int j = threadIdx.x; j < D; j += 256) {
    const float m = mu[(size_t)nn * D + j];
    mubf[(size_t)nn * D + j] = (__bf16)m;
    mm += m * m;
    const float a0 = v[((size_t)nn * 4 + 0) * D + j];
    const float a1 = v[((size_t)nn * 4 + 1) * D + j];
    const float a2 = v[((size_t)nn * 4 + 2) * D + j];
    const float a3 = v[((size_t)nn * 4 + 3) * D + j];
    vT[((size_t)0 * N + nn) * D + j] = (__bf16)a0;
    vT[((size_t)1 * N + nn) * D + j] = (__bf16)a1;
    vT[((size_t)2 * N + nn) * D + j] = (__bf16)a2;
    vT[((size_t)3 * N + nn) * D + j] = (__bf16)a3;
    vm0 += a0 * m; vm1 += a1 * m; vm2 += a2 * m; vm3 += a3 * m;
  }
  red[0][threadIdx.x] = mm;
  red[1][threadIdx.x] = vm0; red[2][threadIdx.x] = vm1;
  red[3][threadIdx.x] = vm2; red[4][threadIdx.x] = vm3;
  __syncthreads();
  for (int s = 128; s > 0; s >>= 1) {
    if (threadIdx.x < s)
      for (int t = 0; t < 5; ++t) red[t][threadIdx.x] += red[t][threadIdx.x + s];
    __syncthreads();
  }
  if (threadIdx.x == 0) {
    mumu[nn] = red[0][0];
    vmu[0 * N + nn] = red[1][0]; vmu[1 * N + nn] = red[2][0];
    vmu[2 * N + nn] = red[3][0]; vmu[3 * N + nn] = red[4][0];
  }
}

// --- column sum / sumsq partials (coalesced, atomic combine) ---------------
__global__ __launch_bounds__(256) void col_stats(const float* __restrict__ h,
                                                 float* __restrict__ sum,
                                                 float* __restrict__ sumsq,
                                                 int N, int rowsPerBlock) {
  const int c = blockIdx.x * 256 + threadIdx.x;
  const int r0 = blockIdx.y * rowsPerBlock;
  float s = 0.0f, s2 = 0.0f;
  for (int r = r0; r < r0 + rowsPerBlock; ++r) {
    const float v = h[(size_t)r * N + c];
    s += v; s2 += v * v;
  }
  atomicAdd(&sum[c], s);
  atomicAdd(&sumsq[c], s2);
}

// --- BN apply + bf16 convert + per-row ||h||^2 (layer-2 input) -------------
__global__ __launch_bounds__(256) void bn_apply(
    const float* __restrict__ h, const float* __restrict__ sum,
    const float* __restrict__ sq, const float* __restrict__ g,
    const float* __restrict__ be, __bf16* __restrict__ hbf,
    float* __restrict__ xx2, int Bsz, int N) {
  const int row = blockIdx.x;
  __shared__ float red[256];
  const float invB = 1.0f / (float)Bsz;
  float acc = 0.0f;
  for (int c = threadIdx.x; c < N; c += 256) {
    const float m = sum[c] * invB;
    const float var = sq[c] * invB - m * m;
    const float y = (h[(size_t)row * N + c] - m) * rsqrtf(var + BN_EPS) * g[c] + be[c];
    hbf[(size_t)row * N + c] = (__bf16)y;
    acc += y * y;
  }
  red[threadIdx.x] = acc; __syncthreads();
  for (int s = 128; s > 0; s >>= 1) {
    if (threadIdx.x < s) red[threadIdx.x] += red[threadIdx.x + s];
    __syncthreads();
  }
  if (threadIdx.x == 0) xx2[row] = red[0];
}

// --- final: BN2 + residual -------------------------------------------------
__global__ __launch_bounds__(256) void final_bn_res(
    const float* __restrict__ h2, const float* __restrict__ sum,
    const float* __restrict__ sq, const float* __restrict__ g,
    const float* __restrict__ be, const float* __restrict__ x,
    float* __restrict__ out, int Bsz, int N) {
  const size_t i = (size_t)blockIdx.x * 256 + threadIdx.x;
  if (i >= (size_t)Bsz * N) return;
  const int c = (int)(i % N);
  const float invB = 1.0f / (float)Bsz;
  const float m = sum[c] * invB;
  const float var = sq[c] * invB - m * m;
  out[i] = (h2[i] - m) * rsqrtf(var + BN_EPS) * g[c] + be[c] + x[i];
}

// ---------------------------------------------------------------------------
extern "C" void kernel_launch(void* const* d_in, const int* in_sizes, int n_in,
                              void* d_out, int out_size, void* d_ws, size_t ws_size,
                              hipStream_t stream) {
  const float* x   = (const float*)d_in[0];
  const float* mu1 = (const float*)d_in[1];
  const float* la1 = (const float*)d_in[2];
  const float* v1  = (const float*)d_in[3];
  const float* g1  = (const float*)d_in[4];
  const float* b1  = (const float*)d_in[5];
  const float* mu2 = (const float*)d_in[6];
  const float* la2 = (const float*)d_in[7];
  const float* v2  = (const float*)d_in[8];
  const float* g2  = (const float*)d_in[9];
  const float* b2  = (const float*)d_in[10];

  const int N = in_sizes[2];               // lam1: (n,)
  const int D = in_sizes[1] / N;           // mu1: (n,d)
  const int B = in_sizes[0] / D;           // x: (b,d)

  // workspace carve-out (256B aligned)
  char* ws = (char*)d_ws;
  size_t off = 0;
  auto carve = [&](size_t bytes) -> void* {
    void* p = ws + off;
    off = (off + bytes + 255) & ~(size_t)255;
    return p;
  };
  __bf16* Xbf   = (__bf16*)carve((size_t)B * D * 2);
  __bf16* Hbf   = (__bf16*)carve((size_t)B * N * 2);
  __bf16* Mu1bf = (__bf16*)carve((size_t)N * D * 2);
  __bf16* Mu2bf = (__bf16*)carve((size_t)N * N * 2);
  __bf16* V1T   = (__bf16*)carve((size_t)4 * N * D * 2);
  __bf16* V2T   = (__bf16*)carve((size_t)4 * N * N * 2);
  float*  h1    = (float*)carve((size_t)B * N * 4);
  float*  h2    = (float*)carve((size_t)B * N * 4);
  float*  xx1   = (float*)carve((size_t)B * 4);
  float*  xx2   = (float*)carve((size_t)B * 4);
  float*  mumu1 = (float*)carve((size_t)N * 4);
  float*  mumu2 = (float*)carve((size_t)N * 4);
  float*  vmu1  = (float*)carve((size_t)4 * N * 4);
  float*  vmu2  = (float*)carve((size_t)4 * N * 4);
  float*  stats = (float*)carve((size_t)4 * N * 4);  // s1sum,s1sq,s2sum,s2sq
  float* s1sum = stats, *s1sq = stats + N, *s2sum = stats + 2 * N, *s2sq = stats + 3 * N;

  // 1) zero BN stats (ws is not re-poisoned between replays)
  zero_f32<<<(4 * N + 255) / 256, 256, 0, stream>>>(stats, 4 * N);
  // 2) x -> bf16 + ||x||^2 per row
  prep_x<<<B, 256, 0, stream>>>(x, Xbf, xx1, D);
  // 3) parameter prep for both layers
  prep_params<<<N, 256, 0, stream>>>(mu1, v1, Mu1bf, V1T, mumu1, vmu1, N, D);
  prep_params<<<N, 256, 0, stream>>>(mu2, v2, Mu2bf, V2T, mumu2, vmu2, N, N);
  // 4) layer 1 fused HMU (WMMA), 32x16 tile per wave
  hmu_wmma<<<dim3(N / 64, B / 32), 128, 0, stream>>>(Xbf, Mu1bf, V1T, la1,
                                                     mumu1, vmu1, xx1, h1, D, N);
  // 5) BN1 stats
  col_stats<<<dim3(N / 256, B / 128), 256, 0, stream>>>(h1, s1sum, s1sq, N, 128);
  // 6) BN1 apply -> bf16 H + ||h||^2 per row
  bn_apply<<<B, 256, 0, stream>>>(h1, s1sum, s1sq, g1, b1, Hbf, xx2, B, N);
  // 7) layer 2 fused HMU (WMMA)
  hmu_wmma<<<dim3(N / 64, B / 32), 128, 0, stream>>>(Hbf, Mu2bf, V2T, la2,
                                                     mumu2, vmu2, xx2, h2, N, N);
  // 8) BN2 stats
  col_stats<<<dim3(N / 256, B / 128), 256, 0, stream>>>(h2, s2sum, s2sq, N, 128);
  // 9) BN2 apply + residual
  final_bn_res<<<(int)(((size_t)B * N + 255) / 256), 256, 0, stream>>>(
      h2, s2sum, s2sq, g2, b2, x, (float*)d_out, B, N);
}
